// LMModel_LSTM_4406636446325
// MI455X (gfx1250) — compile-verified, matrix operand order checked by reading, not run on hardware
//
#include <hip/hip_runtime.h>
#include <hip/hip_bf16.h>

#define NVOC    32000
#define DIM     256
#define HID     256
#define NLAYERS 4
#define SEQ     256
#define BATCH   32
#define G4H     (4 * HID)          // 1024
#define SB      (SEQ * BATCH)      // 8192

typedef __attribute__((ext_vector_type(2)))  float    v2f;
typedef __attribute__((ext_vector_type(8)))  float    v8f;
typedef __attribute__((ext_vector_type(8)))  _Float16 v8h;
typedef __attribute__((ext_vector_type(16))) _Float16 v16h;

// ---- workspace byte offsets ----
#define WS_X0     ((size_t)0)          // SB*DIM f32       =  8,388,608 B
#define WS_HSEQ   ((size_t)8388608)    // 3*SB*HID f32     = 25,165,824 B
#define WS_HTOP16 ((size_t)33554432)   // SB*HID f16       =  4,194,304 B
#define WS_WD16   ((size_t)37748736)   // NVOC*DIM f16     = 16,384,000 B
#define WS_PROG   ((size_t)54132736)   // 4 ints

__device__ __forceinline__ v16h cat8(v8h lo, v8h hi) {
    return __builtin_shufflevector(lo, hi, 0,1,2,3,4,5,6,7,8,9,10,11,12,13,14,15);
}

__device__ __forceinline__ float sigmoidf_(float x) {
    return 1.0f / (1.0f + __expf(-x));
}

// ------------------------------------------------------------------ helpers
__global__ void init_prog_kernel(int* prog) {
    if (threadIdx.x < NLAYERS) prog[threadIdx.x] = 0;
}

__global__ void embed_kernel(const int* __restrict__ ids,
                             const float* __restrict__ emb,
                             float* __restrict__ x0) {
    size_t idx = (size_t)blockIdx.x * blockDim.x + threadIdx.x;   // over SB*DIM
    if (idx >= (size_t)SB * DIM) return;
    int row = (int)(idx >> 8);      // / DIM
    int col = (int)(idx & 255);
    int tok = ids[row];
    x0[idx] = emb[(size_t)tok * DIM + col];
}

__global__ void f32_to_f16_kernel(const float* __restrict__ src,
                                  _Float16* __restrict__ dst, int n) {
    int i = blockIdx.x * blockDim.x + threadIdx.x;
    if (i < n) dst[i] = (_Float16)src[i];
}

// --------------------------------------------------------- recurrent kernel
// grid = NLAYERS blocks of 1024 threads (32 waves), one block per layer,
// pipelined over time with agent-scope progress flags. Exact fp32 WMMA.
__global__ void __launch_bounds__(1024) lstm_kernel(
    const float* __restrict__ x0,     // [SEQ,BATCH,DIM]
    const float* __restrict__ Wx,     // [L,4H,DIM]
    const float* __restrict__ Wh,     // [L,4H,HID]
    const float* __restrict__ bx,     // [L,4H]
    const float* __restrict__ bh,     // [L,4H]
    float*       __restrict__ hseq,   // [3,SEQ,BATCH,HID] (layers 0..2 outputs)
    _Float16*    __restrict__ htop,   // [SEQ,BATCH,HID]   (layer 3 output, f16)
    float*       __restrict__ hn_out, // [L,BATCH,HID]
    float*       __restrict__ cn_out, // [L,BATCH,HID]
    int*         __restrict__ prog)
{
    extern __shared__ char smem_raw[];
    float* zbuf = (float*)smem_raw;            // [BATCH][4H]  = 128 KB
    float* hbuf = zbuf + BATCH * G4H;          // [BATCH][HID] =  32 KB

    const int l    = blockIdx.x;
    const int tid  = threadIdx.x;
    const int wave = tid >> 5;
    const int lane = tid & 31;
    const int ml   = lane & 15;
    const int hi   = lane >> 4;
    const int kb   = hi * 2;                   // K sub-offset for 16x16x4 layout
    const int n0   = wave * 32;                // this wave owns N columns [n0,n0+32)

    const float* wxl = Wx + (size_t)l * G4H * DIM;
    const float* whl = Wh + (size_t)l * G4H * HID;
    const float* bxl = bx + l * G4H;
    const float* bhl = bh + l * G4H;
    const float* xin = (l == 0) ? x0 : (hseq + (size_t)(l - 1) * SB * HID);

    for (int i = tid; i < BATCH * HID; i += 1024) hbuf[i] = 0.0f;
    float creg[8];
    #pragma unroll
    for (int i = 0; i < 8; ++i) creg[i] = 0.0f;
    __syncthreads();

    for (int t = 0; t < SEQ; ++t) {
        if (l > 0) {
            if (tid == 0) {
                while (__hip_atomic_load(&prog[l - 1], __ATOMIC_ACQUIRE,
                                         __HIP_MEMORY_SCOPE_AGENT) <= t) {
                    __builtin_amdgcn_s_sleep(2);
                }
            }
            __syncthreads();
            __builtin_amdgcn_fence(__ATOMIC_ACQUIRE, "agent");
        }

        v8f a00 = {}, a01 = {}, a10 = {}, a11 = {};

        // ---- z += x_t @ Wx^T  (K = DIM, exact fp32 WMMA 16x16x4) ----
        const float* xrow0 = xin + ((size_t)t * BATCH + ml) * DIM;
        const float* xrow1 = xrow0 + (size_t)16 * DIM;
        const float* w0 = wxl + (size_t)(n0 + ml) * DIM;
        const float* w1 = wxl + (size_t)(n0 + 16 + ml) * DIM;
        #pragma unroll 4
        for (int k = 0; k < DIM; k += 4) {
            v2f av0 = *(const v2f*)(xrow0 + k + kb);
            v2f av1 = *(const v2f*)(xrow1 + k + kb);
            v2f bv0 = *(const v2f*)(w0 + k + kb);
            v2f bv1 = *(const v2f*)(w1 + k + kb);
            a00 = __builtin_amdgcn_wmma_f32_16x16x4_f32(false, av0, false, bv0, (short)0, a00, false, false);
            a01 = __builtin_amdgcn_wmma_f32_16x16x4_f32(false, av0, false, bv1, (short)0, a01, false, false);
            a10 = __builtin_amdgcn_wmma_f32_16x16x4_f32(false, av1, false, bv0, (short)0, a10, false, false);
            a11 = __builtin_amdgcn_wmma_f32_16x16x4_f32(false, av1, false, bv1, (short)0, a11, false, false);
        }

        // ---- z += h_{t-1} @ Wh^T  (h from LDS) ----
        const float* hrow0 = hbuf + ml * HID;
        const float* hrow1 = hbuf + (16 + ml) * HID;
        const float* v0 = whl + (size_t)(n0 + ml) * HID;
        const float* v1 = whl + (size_t)(n0 + 16 + ml) * HID;
        #pragma unroll 4
        for (int k = 0; k < HID; k += 4) {
            v2f av0 = *(const v2f*)(hrow0 + k + kb);
            v2f av1 = *(const v2f*)(hrow1 + k + kb);
            v2f bv0 = *(const v2f*)(v0 + k + kb);
            v2f bv1 = *(const v2f*)(v1 + k + kb);
            a00 = __builtin_amdgcn_wmma_f32_16x16x4_f32(false, av0, false, bv0, (short)0, a00, false, false);
            a01 = __builtin_amdgcn_wmma_f32_16x16x4_f32(false, av0, false, bv1, (short)0, a01, false, false);
            a10 = __builtin_amdgcn_wmma_f32_16x16x4_f32(false, av1, false, bv0, (short)0, a10, false, false);
            a11 = __builtin_amdgcn_wmma_f32_16x16x4_f32(false, av1, false, bv1, (short)0, a11, false, false);
        }

        // ---- scatter accumulator tiles into z LDS ----
        #pragma unroll
        for (int j = 0; j < 8; ++j) {
            int mr = j + hi * 8;                      // C layout: VGPR j -> M=j (+8 for hi lanes)
            zbuf[mr * G4H + n0 + ml]             = a00[j];
            zbuf[mr * G4H + n0 + 16 + ml]        = a01[j];
            zbuf[(mr + 16) * G4H + n0 + ml]      = a10[j];
            zbuf[(mr + 16) * G4H + n0 + 16 + ml] = a11[j];
        }
        __syncthreads();

        // ---- gates + state update (each thread owns 8 fixed (b,j) cells) ----
        #pragma unroll
        for (int i = 0; i < 8; ++i) {
            int e = tid + i * 1024;      // 0..8191
            int b = e >> 8;
            int j = e & 255;
            const float* zr = zbuf + b * G4H;
            float zi = zr[j]       + bxl[j]       + bhl[j];
            float zf = zr[256 + j] + bxl[256 + j] + bhl[256 + j];
            float zo = zr[512 + j] + bxl[512 + j] + bhl[512 + j];
            float zg = zr[768 + j] + bxl[768 + j] + bhl[768 + j];
            float cn = sigmoidf_(zf) * creg[i] + sigmoidf_(zi) * tanhf(zg);
            float hn = sigmoidf_(zo) * tanhf(cn);
            creg[i] = cn;
            hbuf[e] = hn;
            size_t gidx = (size_t)t * (BATCH * HID) + e;
            if (l < NLAYERS - 1) hseq[(size_t)l * SB * HID + gidx] = hn;
            else                 htop[gidx] = (_Float16)hn;
            if (t == SEQ - 1) {
                hn_out[l * BATCH * HID + e] = hn;
                cn_out[l * BATCH * HID + e] = cn;
            }
        }
        __syncthreads();

        if (tid == 0) {
            __builtin_amdgcn_fence(__ATOMIC_RELEASE, "agent");
            __hip_atomic_store(&prog[l], t + 1, __ATOMIC_RELEASE,
                               __HIP_MEMORY_SCOPE_AGENT);
        }
    }
}

// ------------------------------------------------------- output projection
// logits[8192, 32000] = h_top(f16) @ Wd(f16)^T + bd, f32 accumulate.
// Block = 8 waves sharing one 64-col N-strip (B panel async-staged in LDS);
// wave w owns M-tile m0 = mb*128 + w*16. Grid: 64 * 500 blocks.
__global__ void __launch_bounds__(256) proj_kernel(
    const _Float16* __restrict__ h16,   // [SB, DIM]
    const _Float16* __restrict__ wd16,  // [NVOC, DIM]
    const float*    __restrict__ bd,    // [NVOC]
    float*          __restrict__ logits)// [SB, NVOC]
{
    __shared__ _Float16 Bsh[64 * DIM];            // 32 KB B panel

    const int tid  = threadIdx.x;
    const int wave = tid >> 5;
    const int lane = tid & 31;
    const int ml   = lane & 15;
    const int hi   = lane >> 4;

    const int mb = blockIdx.x / 500;              // 0..63
    const int nb = blockIdx.x % 500;              // 0..499
    const int m0 = mb * 128 + wave * 16;
    const int n0 = nb * 64;

    // ---- async-stage the 64x256 f16 B panel (Wd rows n0..n0+63) into LDS ----
    // 32 KB = 2048 x 16B chunks; 256 threads x 8 chunks, each via the CDNA5
    // async data mover path (tracked by ASYNCcnt, no VGPR round-trip).
    #pragma unroll
    for (int i = 0; i < 8; ++i) {
        int c = tid + i * 256;                    // chunk id 0..2047
        int row = c >> 5;                         // 32 chunks per 512B row
        int cin = c & 31;
        const _Float16* gsrc = wd16 + (size_t)(n0 + row) * DIM + cin * 8;
        unsigned lds_off = (unsigned)(size_t)&Bsh[row * DIM + cin * 8];
        asm volatile("global_load_async_to_lds_b128 %0, %1, off"
                     :: "v"(lds_off), "v"(gsrc) : "memory");
    }
    asm volatile("s_wait_asynccnt 0" ::: "memory");
    __syncthreads();

    v8f acc[4] = {};
    const _Float16* arow = h16 + (size_t)(m0 + ml) * DIM;

    #pragma unroll
    for (int k = 0; k < DIM; k += 32) {
        const int o1 = k + hi * 8;
        const int o2 = k + 16 + hi * 8;
        v16h a = cat8(*(const v8h*)(arow + o1), *(const v8h*)(arow + o2));
        #pragma unroll
        for (int nt = 0; nt < 4; ++nt) {
            const _Float16* br = &Bsh[(nt * 16 + ml) * DIM];
            v16h b = cat8(*(const v8h*)(br + o1), *(const v8h*)(br + o2));
            acc[nt] = __builtin_amdgcn_wmma_f32_16x16x32_f16(
                false, a, false, b, (short)0, acc[nt], false, false);
        }
    }

    #pragma unroll
    for (int nt = 0; nt < 4; ++nt) {
        int col = n0 + nt * 16 + ml;
        float bias = bd[col];
        #pragma unroll
        for (int j = 0; j < 8; ++j) {
            int row = m0 + j + hi * 8;
            logits[(size_t)row * NVOC + col] = acc[nt][j] + bias;
        }
    }
}

// ----------------------------------------------------------------- launcher
extern "C" void kernel_launch(void* const* d_in, const int* in_sizes, int n_in,
                              void* d_out, int out_size, void* d_ws, size_t ws_size,
                              hipStream_t stream) {
    (void)in_sizes; (void)n_in; (void)out_size; (void)ws_size;
    const int*   ids = (const int*)  d_in[0];
    const float* emb = (const float*)d_in[1];
    const float* Wx  = (const float*)d_in[2];
    const float* Wh  = (const float*)d_in[3];
    const float* bx  = (const float*)d_in[4];
    const float* bh  = (const float*)d_in[5];
    const float* Wd  = (const float*)d_in[6];
    const float* bd  = (const float*)d_in[7];

    char* ws = (char*)d_ws;
    float*    x0   = (float*)   (ws + WS_X0);
    float*    hseq = (float*)   (ws + WS_HSEQ);
    _Float16* htop = (_Float16*)(ws + WS_HTOP16);
    _Float16* wd16 = (_Float16*)(ws + WS_WD16);
    int*      prog = (int*)     (ws + WS_PROG);

    float* logits = (float*)d_out;
    float* hn = logits + (size_t)SB * NVOC;
    float* cn = hn + NLAYERS * BATCH * HID;

    init_prog_kernel<<<1, 32, 0, stream>>>(prog);
    embed_kernel<<<(SB * DIM) / 256, 256, 0, stream>>>(ids, emb, x0);
    f32_to_f16_kernel<<<(NVOC * DIM) / 256, 256, 0, stream>>>(Wd, wd16, NVOC * DIM);

    const size_t lds_bytes = (size_t)(BATCH * G4H + BATCH * HID) * sizeof(float); // 160 KB
    lstm_kernel<<<NLAYERS, 1024, lds_bytes, stream>>>(
        x0, Wx, Wh, bx, bh, hseq, htop, hn, cn, prog);

    proj_kernel<<<64 * 500, 256, 0, stream>>>(htop, wd16, bd, logits);
}
